// GSE3Res_18743237279830
// MI455X (gfx1250) — matrix operand order repaired and checked by reading, not compiled, not run on hardware
//
#include <hip/hip_runtime.h>

typedef __attribute__((ext_vector_type(16))) _Float16 v16h;
typedef __attribute__((ext_vector_type(8)))  float    v8f;

#define NN 20000
#define NE 320000
#define ED1 5       // edge_dim + 1
#define TILE 16
#define WPB 4       // waves per block (wave32)

struct RadialP {     // raw f32 params (prep input)
  const float *w1,*b1,*g1,*be1,*w2,*b2,*g2,*be2,*w3,*b3;
};
struct RadialF {     // preprocessed: f16 fragment-linear weights + folded biases
  const _Float16* fb;   // [4+NCH][32 lanes][16 halfs]
  const float* b2;      // [32]  (b2 + w2*beta1 folded)
  const float* b3;      // [fan] (b3 + w3*beta2 folded)
};

struct EdgeArgs {
  const float *f0,*f1,*efeat;
  const int *esrc,*edst;
  const float *basis00,*basis01,*basis10,*basis11;
  RadialF rv[4], rk[4];
  const float *q0b,*q1b;
  float *msgv0,*msgv1,*ebuf;
};

struct PrepArgs {
  RadialP r[8];          // v00,v01,v10,v11,k00,k01,k10,k11
  _Float16* frag;
  float* b2p;            // [8][32]
  float* b3p;            // concatenated fans
  int fragoff[8];        // in halfs
  int b3off[8];
  int nch3[8];           // fan/16 per radial
};

__device__ __forceinline__ void wave_lds_sync(){
  asm volatile("s_wait_dscnt 0" ::: "memory");
  __builtin_amdgcn_wave_barrier();
}

__device__ __forceinline__ v8f wmma16(v16h a, v16h b, v8f c){
  return __builtin_amdgcn_wmma_f32_16x16x32_f16(false, a, false, b, (short)0, c, false, false);
}

// one aligned 32B load per B fragment (pre-swizzled by prep kernel)
__device__ __forceinline__ v16h ldfrag(const _Float16* fb, int c, int lane){
  return *(const v16h*)(fb + ((c*32 + lane) << 4));
}

// raw C (x already includes bias) -> LDS -> row LN stats -> normalized f16 A frag
__device__ __forceinline__ v16h ln_relu_t(v8f c0, v8f c1,
    float* shx /*[16][32]*/, float* shstat /*[16][2]*/, int lane){
  const int n = lane & 15, hi = lane >> 4, m = lane & 15, hb = hi*8;
#pragma unroll
  for (int v = 0; v < 8; ++v){
    shx[(v + hi*8)*32 + n]      = c0[v];
    shx[(v + hi*8)*32 + 16 + n] = c1[v];
  }
  wave_lds_sync();
  {
    int r = lane >> 1, hf = (lane & 1)*16;
    const float4* px = (const float4*)(shx + r*32 + hf);
    float s = 0.f, s2 = 0.f;
#pragma unroll
    for (int u = 0; u < 4; ++u){
      float4 x = px[u];
      s  += x.x + x.y + x.z + x.w;
      s2 += x.x*x.x + x.y*x.y + x.z*x.z + x.w*x.w;
    }
    s  += __shfl_xor(s,  1, 32);
    s2 += __shfl_xor(s2, 1, 32);
    float mu  = s * (1.0f/32.0f);
    float var = s2 * (1.0f/32.0f) - mu*mu;
    float rs  = rsqrtf(var + 1e-5f);
    if ((lane & 1) == 0){ shstat[r*2] = mu; shstat[r*2+1] = rs; }
  }
  wave_lds_sync();
  float mu = shstat[m*2], rs = shstat[m*2+1];
  float xv[16];
  {
    const float4* p0 = (const float4*)(shx + m*32 + hi*16);
    const float4* p1 = (const float4*)(shx + m*32 + hi*16 + 8);
    *(float4*)&xv[0]  = p0[0]; *(float4*)&xv[4]  = p0[1];
    *(float4*)&xv[8]  = p1[0]; *(float4*)&xv[12] = p1[1];
  }
  v16h a;
#pragma unroll
  for (int i = 0; i < 16; ++i)
    a[i] = (_Float16)fmaxf((xv[i] - mu) * rs, 0.0f);
  return a;
}

// layers 1+2 (bias1 folded into w1's K=5 row; gamma/beta folded by prep) -> A3 frag
__device__ __forceinline__ v16h mlp_front(const RadialF& rf, v16h aX,
    float* shx, float* shstat, int lane){
  v8f c0 = {}, c1 = {};
  c0 = wmma16(aX, ldfrag(rf.fb, 0, lane), c0);
  c1 = wmma16(aX, ldfrag(rf.fb, 1, lane), c1);
  v16h a2 = ln_relu_t(c0, c1, shx, shstat, lane);
  v8f d0 = {}, d1 = {};
  d0 = wmma16(a2, ldfrag(rf.fb, 2, lane), d0);
  d1 = wmma16(a2, ldfrag(rf.fb, 3, lane), d1);
  const int nn = lane & 15;
  float bb0 = rf.b2[nn], bb1 = rf.b2[16 + nn];
#pragma unroll
  for (int v = 0; v < 8; ++v){ d0[v] += bb0; d1[v] += bb1; }
  return ln_relu_t(d0, d1, shx, shstat, lane);
}

// One (di,do) pair: layer3 streamed in 16-col chunks; chunk -> LDS transpose ->
// per-lane partial over 8 cols -> ds_add into G[e,o,q,f]; then msg += G*basis.
template<int DI, int DO>
__device__ __forceinline__ void pair_accum(const RadialF& rf, v16h aX,
    const float* basis, long tb, float* msg,
    float* shx, float* shstat, float* shc /*[16*17]*/,
    float (*shG)[8][9], const float* shs /*[16][32][4]*/, int lane)
{
  constexpr int NF  = (DI && DO) ? 3 : 1;
  constexpr int QN  = 2*DI + 1;
  constexpr int PN  = 2*DO + 1;
  constexpr int NCH = (NF*32*8)/16;

  float* gp = &shG[0][0][0];
  for (int t = lane; t < TILE*8*9; t += 32) gp[t] = 0.0f;

  v16h a3 = mlp_front(rf, aX, shx, shstat, lane);   // contains syncs

  const int n = lane & 15, hi = lane >> 4, m = lane & 15;

  for (int t = 0; t < NCH; ++t){
    v8f c = {};
    c = wmma16(a3, ldfrag(rf.fb, 4 + t, lane), c);
    float bias = rf.b3[t*16 + n];
#pragma unroll
    for (int v = 0; v < 8; ++v) shc[(v + hi*8)*17 + n] = c[v] + bias;
    wave_lds_sync();

    const int o  = (t*16) / (32*NF);     // all 16 cols of a chunk share one o
    const int cb = t*16 - o*32*NF;
    float part[QN*NF];
#pragma unroll
    for (int i = 0; i < QN*NF; ++i) part[i] = 0.0f;
#pragma unroll
    for (int j = 0; j < 8; ++j){
      int col = hi*8 + j;
      int rem = cb + col;
      int cc  = rem / NF;
      int f   = rem % NF;
      float rv = shc[m*17 + col];
      if (DI == 0){
        part[f] += rv * shs[(m*32 + cc)*4 + 3];
      } else {
        float4 sv = *(const float4*)&shs[(m*32 + cc)*4];
        part[0*NF + f] += rv * sv.x;
        part[1*NF + f] += rv * sv.y;
        part[2*NF + f] += rv * sv.z;
      }
    }
#pragma unroll
    for (int i = 0; i < QN*NF; ++i){
      int q = i / NF, f = i % NF;
      atomicAdd(&shG[m][o][q*3 + f], part[i]);    // ds_add_f32, 2-way max
    }
  }
  wave_lds_sync();

  long e  = tb + m;
  long ec = (e < NE) ? e : (NE - 1);
  const float* bp = basis + ec * (long)(PN*QN*NF);
#pragma unroll
  for (int lo = 0; lo < 4; ++lo){
    int o = hi*4 + lo;
#pragma unroll
    for (int q = 0; q < QN; ++q)
#pragma unroll
      for (int f = 0; f < NF; ++f){
        float g = shG[m][o][q*3 + f];
#pragma unroll
        for (int p = 0; p < PN; ++p)
          msg[lo*PN + p] += g * bp[(p*QN + q)*NF + f];
      }
  }
}

__global__ __launch_bounds__(WPB*32)
void edge_kernel(EdgeArgs A){
  __shared__ float sh_x[WPB][TILE][32];        // 8 KB
  __shared__ float sh_stat[WPB][TILE][2];      // 0.5 KB
  __shared__ float sh_c[WPB][TILE*17];         // 4.25 KB (stride-17 pad)
  __shared__ float sh_G[WPB][TILE][8][9];      // 18 KB
  __shared__ float sh_s[WPB][TILE][32][4];     // 32 KB  {f1x,f1y,f1z,f0}

  const int lane = threadIdx.x & 31;
  const int wv   = threadIdx.x >> 5;
  const long tile = (long)blockIdx.x * WPB + wv;
  const long tb   = tile * TILE;
  const int m = lane & 15, hi = lane >> 4;
  const int hb8 = hi * 8;

  long e_m   = tb + m;
  long e_m_c = (e_m < NE) ? e_m : (NE - 1);

  // stage gathered source features (edge_src gather; f0/f1 are L2-resident)
  {
    int r = lane >> 1, hf = (lane & 1)*16;
    long ee = tb + r; if (ee >= NE) ee = NE - 1;
    int s = A.esrc[ee];
    const float* p0 = A.f0 + (long)s*32 + hf;
    const float* p1 = A.f1 + ((long)s*32 + hf)*3;
#pragma unroll
    for (int c2 = 0; c2 < 16; ++c2){
      float4 v;
      v.x = p1[c2*3+0]; v.y = p1[c2*3+1]; v.z = p1[c2*3+2]; v.w = p0[c2];
      *(float4*)&sh_s[wv][r][hf + c2][0] = v;
    }
  }
  wave_lds_sync();

  // A_x fragment: edge_feat [16 x 5] zero-padded to K=32, K=5 carries 1.0 (bias row)
  v16h aX;
  {
    const float* efr = A.efeat + e_m_c * ED1;
#pragma unroll
    for (int i = 0; i < 8; ++i){
      int k = hi*16 + i;
      aX[i]     = (k < ED1) ? (_Float16)efr[k] : ((k == ED1) ? (_Float16)1.0f : (_Float16)0.0f);
      aX[8 + i] = (_Float16)0.0f;
    }
  }

  int dst = A.edst[e_m_c];
  float eacc0 = 0.f, eacc1 = 0.f;
  float* shx = &sh_x[wv][0][0];
  float* shst = &sh_stat[wv][0][0];
  float* shc = &sh_c[wv][0];
  const float* shs = &sh_s[wv][0][0][0];

  // ---------- projection V ----------
  {
    float msg0[4] = {0,0,0,0};
    pair_accum<0,0>(A.rv[0], aX, A.basis00, tb, msg0, shx, shst, shc, sh_G[wv], shs, lane);
    pair_accum<1,0>(A.rv[2], aX, A.basis10, tb, msg0, shx, shst, shc, sh_G[wv], shs, lane);
    if (e_m < NE){
#pragma unroll
      for (int lo = 0; lo < 4; ++lo)
        A.msgv0[e_m*8 + hi*4 + lo] = msg0[lo];
    }
    float msg1[12] = {0,0,0,0,0,0,0,0,0,0,0,0};
    pair_accum<0,1>(A.rv[1], aX, A.basis01, tb, msg1, shx, shst, shc, sh_G[wv], shs, lane);
    pair_accum<1,1>(A.rv[3], aX, A.basis11, tb, msg1, shx, shst, shc, sh_G[wv], shs, lane);
    if (e_m < NE){
#pragma unroll
      for (int lo = 0; lo < 4; ++lo)
#pragma unroll
        for (int p = 0; p < 3; ++p)
          A.msgv1[(e_m*8 + hi*4 + lo)*3 + p] = msg1[lo*3 + p];
    }
  }
  // ---------- projection K (fused attention logits) ----------
  {
    float msg0[4] = {0,0,0,0};
    pair_accum<0,0>(A.rk[0], aX, A.basis00, tb, msg0, shx, shst, shc, sh_G[wv], shs, lane);
    pair_accum<1,0>(A.rk[2], aX, A.basis10, tb, msg0, shx, shst, shc, sh_G[wv], shs, lane);
#pragma unroll
    for (int lo = 0; lo < 4; ++lo){
      int o = hi*4 + lo;
      float v = msg0[lo] * A.q0b[(long)dst*8 + o];
      if (((o >> 1) & 1) == 0) eacc0 += v; else eacc1 += v;
    }
    float msg1[12] = {0,0,0,0,0,0,0,0,0,0,0,0};
    pair_accum<0,1>(A.rk[1], aX, A.basis01, tb, msg1, shx, shst, shc, sh_G[wv], shs, lane);
    pair_accum<1,1>(A.rk[3], aX, A.basis11, tb, msg1, shx, shst, shc, sh_G[wv], shs, lane);
#pragma unroll
    for (int lo = 0; lo < 4; ++lo){
      int o = hi*4 + lo;
      float v = 0.f;
#pragma unroll
      for (int p = 0; p < 3; ++p)
        v += msg1[lo*3 + p] * A.q1b[((long)dst*8 + o)*3 + p];
      if (((o >> 1) & 1) == 0) eacc0 += v; else eacc1 += v;
    }
  }
  if (e_m < NE){
    const float sc = 0.17677669529663687f;  // 1/sqrt(32)
    A.ebuf[e_m*4 + hi*2 + 0] = eacc0 * sc;
    A.ebuf[e_m*4 + hi*2 + 1] = eacc1 * sc;
  }
}

// ---------------- weight preprocessing ----------------
// frag[c][lane][slot] = B[k = (lane>>4)*16 + slot][n = lane&15] for chunk c:
//  c 0..1: padded w1^T (+b1 in K=5 row); c 2..3: w2^T * gamma1; c>=4: w3^T * gamma2
__global__ void prep_frag(PrepArgs P){
  int r = blockIdx.y;
  int i = blockIdx.x * blockDim.x + threadIdx.x;
  int nct = 4 + P.nch3[r];
  if (i >= nct*512) return;
  int c    = i >> 9;
  int lane = (i >> 4) & 31;
  int slot = i & 15;
  int n = lane & 15;
  int k = ((lane >> 4) << 4) + slot;
  const RadialP& rp = P.r[r];
  float val;
  if (c < 2){
    int row = c*16 + n;
    val = (k < ED1) ? rp.w1[row*ED1 + k] : ((k == ED1) ? rp.b1[row] : 0.0f);
  } else if (c < 4){
    int row = (c - 2)*16 + n;
    val = rp.w2[row*32 + k] * rp.g1[k];
  } else {
    int row = (c - 4)*16 + n;
    val = rp.w3[row*32 + k] * rp.g2[k];
  }
  P.frag[P.fragoff[r] + i] = (_Float16)val;
}

__global__ void prep_bias(PrepArgs P){
  int i = blockIdx.x * blockDim.x + threadIdx.x;
  if (i < 256){
    int r = i >> 5, n = i & 31;
    const RadialP& rp = P.r[r];
    float a = rp.b2[n];
    for (int k = 0; k < 32; ++k) a += rp.be1[k] * rp.w2[n*32 + k];
    P.b2p[i] = a;
  } else {
    int j = i - 256;
    int base = 0, r = 0;
    for (r = 0; r < 8; ++r){
      int fan = P.nch3[r]*16;
      if (j < base + fan) break;
      base += fan;
    }
    if (r >= 8) return;
    int n = j - base;
    const RadialP& rp = P.r[r];
    float a = rp.b3[n];
    for (int k = 0; k < 32; ++k) a += rp.be2[k] * rp.w3[n*32 + k];
    P.b3p[P.b3off[r] + n] = a;
  }
}

// ---------------- auxiliary kernels ----------------

__global__ void init_kernel(float* den, unsigned* mk, float* agg0, float* agg1){
  int i = blockIdx.x * blockDim.x + threadIdx.x;
  if (i < NN*4){ den[i] = 0.f; mk[i] = 0u; }
  if (i < NN*8)  agg0[i] = 0.f;
  if (i < NN*24) agg1[i] = 0.f;
}

__global__ void q_kernel(const float* f0, const float* f1,
                         const float* pq0, const float* pq1,
                         float* q0b, float* q1b){
  int i = blockIdx.x * blockDim.x + threadIdx.x;
  if (i >= NN*8) return;
  int nn = i >> 3, o = i & 7;
  const float* w0 = pq0 + o*32;
  const float* w1 = pq1 + o*32;
  const float* x0 = f0 + (long)nn*32;
  const float* x1 = f1 + (long)nn*96;
  float a = 0.f, b0 = 0.f, b1 = 0.f, b2 = 0.f;
  for (int t = 0; t < 32; ++t){
    a += w0[t] * x0[t];
    float wt = w1[t];
    b0 += wt * x1[t*3+0];
    b1 += wt * x1[t*3+1];
    b2 += wt * x1[t*3+2];
  }
  q0b[i] = a;
  q1b[i*3+0] = b0; q1b[i*3+1] = b1; q1b[i*3+2] = b2;
}

__global__ void emax_kernel(const float* ebuf, const int* edst, unsigned* mk){
  int i = blockIdx.x * blockDim.x + threadIdx.x;
  if (i >= NE*4) return;
  int e = i >> 2, h = i & 3;
  unsigned u = __float_as_uint(ebuf[i]);
  unsigned k = (u & 0x80000000u) ? ~u : (u | 0x80000000u);
  atomicMax(&mk[(long)edst[e]*4 + h], k);
}

__global__ void esoft_kernel(const float* ebuf, const int* edst, const unsigned* mk,
                             float* eebuf, float* den){
  int i = blockIdx.x * blockDim.x + threadIdx.x;
  if (i >= NE*4) return;
  int e = i >> 2, h = i & 3;
  unsigned k = mk[(long)edst[e]*4 + h];
  unsigned u = (k & 0x80000000u) ? (k & 0x7fffffffu) : ~k;
  float mx = __uint_as_float(u);
  float ee = expf(ebuf[i] - mx);
  eebuf[i] = ee;
  atomicAdd(&den[(long)edst[e]*4 + h], ee);
}

__global__ void agg_kernel(const float* eebuf, const float* den, const int* edst,
                           const float* msgv0, const float* msgv1,
                           float* agg0, float* agg1){
  long i = (long)blockIdx.x * blockDim.x + threadIdx.x;
  if (i >= (long)NE*32) return;
  long e = i >> 5; int s = (int)(i & 31);
  int dst = edst[e];
  int o = (s < 8) ? s : (s - 8) / 3;
  int h = o >> 1;
  float alpha = eebuf[e*4 + h] / den[(long)dst*4 + h];
  if (s < 8)
    atomicAdd(&agg0[(long)dst*8 + o], alpha * msgv0[e*8 + o]);
  else {
    int p = (s - 8) % 3;
    atomicAdd(&agg1[((long)dst*8 + o)*3 + p], alpha * msgv1[(e*8 + o)*3 + p]);
  }
}

__global__ void out_kernel(const float* f0, const float* f1,
                           const float* agg0, const float* agg1,
                           const float* proj0, const float* proj1, float* out){
  long i = (long)blockIdx.x * blockDim.x + threadIdx.x;
  if (i >= (long)NN*128) return;
  long nn = i >> 7; int j = (int)(i & 127);
  if (j < 32){
    int o = j;
    const float* w = proj0 + o*40;
    float acc = 0.f;
    for (int t = 0; t < 8;  ++t) acc += w[t]   * agg0[nn*8 + t];
    for (int t = 0; t < 32; ++t) acc += w[8+t] * f0[nn*32 + t];
    out[nn*32 + o] = acc;
  } else {
    int jj = j - 32, o = jj/3, p = jj%3;
    const float* w = proj1 + o*40;
    float acc = 0.f;
    for (int t = 0; t < 8;  ++t) acc += w[t]   * agg1[(nn*8 + t)*3 + p];
    for (int t = 0; t < 32; ++t) acc += w[8+t] * f1[(nn*32 + t)*3 + p];
    out[(long)NN*32 + (nn*32 + o)*3 + p] = acc;
  }
}

static RadialP mk_radial(void* const* d_in, int b){
  RadialP r;
  r.w1 =(const float*)d_in[b+0]; r.b1 =(const float*)d_in[b+1];
  r.g1 =(const float*)d_in[b+2]; r.be1=(const float*)d_in[b+3];
  r.w2 =(const float*)d_in[b+4]; r.b2 =(const float*)d_in[b+5];
  r.g2 =(const float*)d_in[b+6]; r.be2=(const float*)d_in[b+7];
  r.w3 =(const float*)d_in[b+8]; r.b3 =(const float*)d_in[b+9];
  return r;
}

extern "C" void kernel_launch(void* const* d_in, const int* in_sizes, int n_in,
                              void* d_out, int out_size, void* d_ws, size_t ws_size,
                              hipStream_t stream){
  (void)in_sizes; (void)n_in; (void)out_size; (void)ws_size;
  // setup_inputs() insertion order:
  // 0:f0 1:f1 2:edge_src 3:edge_dst 4:edge_feat 5..8:basis_00,01,10,11
  // params: v{00,01,10,11}(w1,b1,g1,be1,w2,b2,g2,be2,w3,b3)->9..48, k->49..88,
  //         q{0,1}->89,90, proj{0,1}->91,92
  const float* f0    = (const float*)d_in[0];
  const float* f1    = (const float*)d_in[1];
  const int*   esrc  = (const int*)d_in[2];
  const int*   edst  = (const int*)d_in[3];
  const float* efeat = (const float*)d_in[4];
  const float* pq0   = (const float*)d_in[89];
  const float* pq1   = (const float*)d_in[90];
  const float* proj0 = (const float*)d_in[91];
  const float* proj1 = (const float*)d_in[92];

  PrepArgs P;
  static const int nch3[8]   = {16,16,16,48, 16,16,16,48};
  int fragoff[8], b3off[8];
  int fo = 0, b3o = 0;
  for (int r = 0; r < 8; ++r){
    P.r[r] = mk_radial(d_in, (r < 4 ? 9 : 49) + 10*(r & 3));
    P.nch3[r] = nch3[r];
    fragoff[r] = fo; P.fragoff[r] = fo; fo  += (4 + nch3[r]) * 512;
    b3off[r]  = b3o; P.b3off[r]  = b3o; b3o += nch3[r] * 16;
  }
  const int totfragh = fo;      // 114688 halfs
  const int totb3    = b3o;     // 3072 floats

  // workspace layout
  _Float16* frag = (_Float16*)d_ws;
  float* w = (float*)d_ws + (totfragh >> 1);   // halfs -> float offset
  size_t off = 0;
  float* b2p   = w + off; off += 256;
  float* b3p   = w + off; off += (size_t)totb3;
  float* q0b   = w + off; off += (size_t)NN*8;
  float* q1b   = w + off; off += (size_t)NN*24;
  float* msgv0 = w + off; off += (size_t)NE*8;
  float* msgv1 = w + off; off += (size_t)NE*24;
  float* ebuf  = w + off; off += (size_t)NE*4;
  float* eebuf = w + off; off += (size_t)NE*4;
  unsigned* mk = (unsigned*)(w + off); off += (size_t)NN*4;
  float* den   = w + off; off += (size_t)NN*4;
  float* agg0  = w + off; off += (size_t)NN*8;
  float* agg1  = w + off; off += (size_t)NN*24;

  P.frag = frag; P.b2p = b2p; P.b3p = b3p;

  EdgeArgs A;
  A.f0 = f0; A.f1 = f1; A.efeat = efeat; A.esrc = esrc; A.edst = edst;
  A.basis00 = (const float*)d_in[5];
  A.basis01 = (const float*)d_in[6];
  A.basis10 = (const float*)d_in[7];
  A.basis11 = (const float*)d_in[8];
  for (int p = 0; p < 4; ++p){
    A.rv[p].fb = frag + fragoff[p];     A.rv[p].b2 = b2p + p*32;     A.rv[p].b3 = b3p + b3off[p];
    A.rk[p].fb = frag + fragoff[4+p];   A.rk[p].b2 = b2p + (4+p)*32; A.rk[p].b3 = b3p + b3off[4+p];
  }
  A.q0b = q0b; A.q1b = q1b; A.msgv0 = msgv0; A.msgv1 = msgv1; A.ebuf = ebuf;

  // 1) preprocess weights into f16 fragment-linear layout (+LN folding)
  prep_frag<<<dim3((52*512 + 255)/256, 8), 256, 0, stream>>>(P);
  prep_bias<<<(256 + totb3 + 255)/256, 256, 0, stream>>>(P);
  // 2) zero accumulators, compute q
  init_kernel<<<(NN*24 + 255)/256, 256, 0, stream>>>(den, mk, agg0, agg1);
  q_kernel<<<(NN*8 + 255)/256, 256, 0, stream>>>(f0, f1, pq0, pq1, q0b, q1b);
  // 3) heavy fused edge kernel (WMMA radial MLPs + basis einsum + logits)
  const int ntiles = (NE + TILE - 1)/TILE;
  const int nblk   = (ntiles + WPB - 1)/WPB;
  edge_kernel<<<nblk, WPB*32, 0, stream>>>(A);
  // 4) segment softmax + aggregation + output projection
  emax_kernel<<<(NE*4 + 255)/256, 256, 0, stream>>>(ebuf, edst, mk);
  esoft_kernel<<<(NE*4 + 255)/256, 256, 0, stream>>>(ebuf, edst, mk, eebuf, den);
  agg_kernel<<<(int)(((long)NE*32 + 255)/256), 256, 0, stream>>>(eebuf, den, edst, msgv0, msgv1, agg0, agg1);
  out_kernel<<<(int)(((long)NN*128 + 255)/256), 256, 0, stream>>>(f0, f1, agg0, agg1, proj0, proj1, (float*)d_out);
}
